// HAN_1219770712439
// MI455X (gfx1250) — compile-verified, matrix operand order checked by reading, not run on hardware
//
#include <hip/hip_runtime.h>
#include <hip/hip_bf16.h>

// ---------------------------------------------------------------------------
// HAN (3x hetero-GAT) + set-transformer forward for MI455X (gfx1250, wave32).
// All dense GEMMs go through v_wmma_f32_16x16x32_f16 with f32 accumulate.
// W is staged into LDS pre-swizzled into the per-lane B-fragment layout so
// each fragment is one aligned 32B LDS read (2x ds_load_b128) instead of 16
// scalar ds_load_u16. Edge softmax = 3-pass atomic segment ops.
//
// d_in layout assumption:
//   0:x_inst 1:x_var 2:x_const
//   3/4 src/dst_ctrl 5/6 call 7/8 i2v 9/10 v2i 11/12 c2i 13/14 iid 15/16 vid
//   params as pytree leaves (sorted dict keys, recursive):
//   17 Wout 18 beta1 19 beta2 20 bout 21 g1 22 g2
//   l1: 23 W_const 24 W_inst 25 W_var 26 Wk 27 ad_c2i 28 ad_call 29 ad_ctrl
//       30 ad_i2v 31 ad_iid 32 ad_v2i 33 ad_vid 34 as_c2i 35 as_call 36 as_ctrl
//       37 as_i2v 38 as_iid 39 as_v2i 40 as_vid 41 b_const 42 b_inst 43 b_var
//       44 bk 45 q
//   l2: 46 W_inst 47 W_var 48 Wk 49 ad_call 50 ad_ctrl 51 ad_i2v 52 ad_iid
//       53 ad_v2i 54 ad_vid 55 as_call 56 as_ctrl 57 as_i2v 58 as_iid
//       59 as_v2i 60 as_vid 61 b_inst 62 b_var 63 bk 64 q
//   l3: 65..83 (same layout as l2)
//   st_i: 84 Wfc 85 Wk 86 Wo 87 Wq 88 Wv 89 bfc 90 bk 91 bo 92 bq 93 bv 94 ip
//   st_v: 95..105 (same layout)
// ---------------------------------------------------------------------------

typedef __attribute__((ext_vector_type(16))) _Float16 v16h;
typedef __attribute__((ext_vector_type(8)))  _Float16 v8h;
typedef __attribute__((ext_vector_type(8)))  float    v8f;

#define INV_SQRT32 0.17677669529663687f
#define MONO_NEGINF 0x007FFFFFu

__device__ __forceinline__ unsigned f2mono(float f) {
    unsigned u = __float_as_uint(f);
    return (u & 0x80000000u) ? ~u : (u | 0x80000000u);
}
__device__ __forceinline__ float mono2f(unsigned m) {
    unsigned u = (m & 0x80000000u) ? (m & 0x7FFFFFFFu) : ~m;
    return __uint_as_float(u);
}

// ---------------------------------------------------------------------------
// WMMA GEMM: C[N,128] = act(A[N,K] @ W[K,128] + bias), K in {32,64,128}.
// mode 0: store; mode 1: store relu; mode 2: atomic column-sum of tanh()
// (semantic-attention score fusion — never materializes the tanh tensor).
// Block = 128 threads = 4 waves; block tile = 64 rows x 128 cols;
// each wave owns 16 rows x 8 WMMA col-tiles.
//
// B-fragment swizzle: element W[k][c] with kb32=k/32, kin=k%32, ct=c/16,
// n=c%16 lives in lane = ((kin>>3)&1)*16 + n, slot = (kin&7) + 8*(kin>=16),
// at Wf[((kb32*8 + ct)*32 + lane)*16 + slot]. Each lane's fragment is then
// 16 contiguous halves (32B aligned) -> 2x ds_load_b128.
// ---------------------------------------------------------------------------
__global__ __launch_bounds__(128)
void k_gemm_wmma(const float* __restrict__ A, const float* __restrict__ W,
                 const float* __restrict__ bias, float* __restrict__ C,
                 float* __restrict__ colsum, int N, int K, int mode)
{
    __shared__ alignas(32) _Float16 As[64 * 128];
    __shared__ alignas(32) _Float16 Wf[128 * 128];   // fragment-swizzled, K*128 used
    const int tid  = threadIdx.x;
    const int wave = tid >> 5;
    const int lane = tid & 31;
    const int rb   = blockIdx.x * 64;

    // prefetch next row-tile of A (lowers to global_prefetch_b8)
    if (tid == 0 && rb + 64 < N)
        __builtin_prefetch(&A[(long long)(rb + 64) * K], 0, 1);

    // stage A (row-major, f16)
    for (int i = tid; i < 64 * K; i += 128) {
        int r = i / K, c = i - r * K;
        int gr = rb + r;
        As[i] = (gr < N) ? (_Float16)A[(long long)gr * K + c] : (_Float16)0.f;
    }
    // stage W swizzled into B-fragment layout
    for (int i = tid; i < K * 128; i += 128) {
        int k = i >> 7, c = i & 127;
        int kb32 = k >> 5, kin = k & 31;
        int ct = c >> 4, n = c & 15;
        int lhalf = (kin >> 3) & 1;                 // 0: K {0..7,16..23}, 1: {8..15,24..31}
        int slot  = (kin & 7) + ((kin >> 4) << 3);  // j + 8*(kin>=16)
        int l     = lhalf * 16 + n;
        Wf[((kb32 * 8 + ct) * 32 + l) * 16 + slot] = (_Float16)W[i];
    }
    __syncthreads();

    v8f acc[8];
#pragma unroll
    for (int t = 0; t < 8; ++t) acc[t] = (v8f){0.f,0.f,0.f,0.f,0.f,0.f,0.f,0.f};

    const int ln15 = lane & 15;          // M index for A-frag
    const int koff = (lane >> 4) * 8;    // K sub-block per lane half

    const int nkb = K >> 5;
    for (int kb32 = 0; kb32 < nkb; ++kb32) {
        const _Float16* ap = &As[(wave * 16 + ln15) * K + (kb32 << 5) + koff];
        const v8h a0 = *(const v8h*)ap;          // 16B aligned
        const v8h a1 = *(const v8h*)(ap + 16);
        v16h af;
#pragma unroll
        for (int j = 0; j < 8; ++j) { af[j] = a0[j]; af[8 + j] = a1[j]; }
#pragma unroll
        for (int ct = 0; ct < 8; ++ct) {
            const v16h bf = *(const v16h*)&Wf[((kb32 * 8 + ct) * 32 + lane) * 16];
            acc[ct] = __builtin_amdgcn_wmma_f32_16x16x32_f16(
                false, af, false, bf, (short)0, acc[ct], false, false);
        }
    }

    const int rgrp = (lane >> 4) * 8;    // C/D: lanes 0-15 -> M 0..7, 16-31 -> M 8..15
    if (mode == 2) {
#pragma unroll
        for (int ct = 0; ct < 8; ++ct) {
            int col = ct * 16 + ln15;
            float bv = bias ? bias[col] : 0.f;
            float part = 0.f;
#pragma unroll
            for (int r = 0; r < 8; ++r) {
                int row = rb + wave * 16 + rgrp + r;
                if (row < N) part += tanhf(acc[ct][r] + bv);
            }
            atomicAdd(&colsum[col], part);
        }
    } else {
#pragma unroll
        for (int ct = 0; ct < 8; ++ct) {
            int col = ct * 16 + ln15;
            float bv = bias ? bias[col] : 0.f;
#pragma unroll
            for (int r = 0; r < 8; ++r) {
                int row = rb + wave * 16 + rgrp + r;
                if (row < N) {
                    float v = acc[ct][r] + bv;
                    if (mode == 1) v = fmaxf(v, 0.f);
                    C[(long long)row * 128 + col] = v;
                }
            }
        }
    }
}

// ---------------------------------------------------------------------------
// Small utility kernels
// ---------------------------------------------------------------------------
__global__ void k_fill_f32(float* p, float v, long long n) {
    long long i = (long long)blockIdx.x * blockDim.x + threadIdx.x;
    if (i < n) p[i] = v;
}
__global__ void k_fill_u32(unsigned* p, unsigned v, long long n) {
    long long i = (long long)blockIdx.x * blockDim.x + threadIdx.x;
    if (i < n) p[i] = v;
}
__global__ void k_map_relu(float* p, long long n) {
    long long i = (long long)blockIdx.x * blockDim.x + threadIdx.x;
    if (i < n) p[i] = fmaxf(p[i], 0.f);
}
__global__ void k_map_elu(const float* __restrict__ x, float* __restrict__ y, long long n) {
    long long i = (long long)blockIdx.x * blockDim.x + threadIdx.x;
    if (i < n) { float v = x[i]; y[i] = (v > 0.f) ? v : (__expf(v) - 1.f); }
}

// s[n,h] = sum_d h[n, h*16+d] * avec[h*16+d]
__global__ void k_node_scores(const float* __restrict__ h, const float* __restrict__ avec,
                              float* __restrict__ s, int Nn)
{
    int idx = blockIdx.x * blockDim.x + threadIdx.x;
    if (idx >= Nn * 8) return;
    int n = idx >> 3, hh = idx & 7;
    const float* hp = h + (long long)n * 128 + hh * 16;
    const float* ap = avec + hh * 16;
    float acc = 0.f;
#pragma unroll
    for (int d = 0; d < 16; ++d) acc += hp[d] * ap[d];
    s[idx] = acc;
}

// Edge pass 1: a = leaky_relu(s_src[src]+s_dst[dst]); segment max via mono-uint
__global__ void k_edge_pass1(const int* __restrict__ src, const int* __restrict__ dst,
                             const float* __restrict__ ssrc, const float* __restrict__ sdst,
                             float* __restrict__ ea, unsigned* __restrict__ amax, int E)
{
    int idx = blockIdx.x * blockDim.x + threadIdx.x;
    if (idx >= E * 8) return;
    int e = idx >> 3, hh = idx & 7;
    float a = ssrc[src[e] * 8 + hh] + sdst[dst[e] * 8 + hh];
    a = (a > 0.f) ? a : 0.1f * a;
    ea[idx] = a;
    atomicMax(&amax[dst[e] * 8 + hh], f2mono(a));
}

// Edge pass 2: e = exp(a - amax[dst]); segment sum of e
__global__ void k_edge_pass2(const int* __restrict__ dst, float* __restrict__ ea,
                             const unsigned* __restrict__ amax, float* __restrict__ denom, int E)
{
    int idx = blockIdx.x * blockDim.x + threadIdx.x;
    if (idx >= E * 8) return;
    int e = idx >> 3, hh = idx & 7;
    int di = dst[e] * 8 + hh;
    float ex = __expf(ea[idx] - mono2f(amax[di]));
    ea[idx] = ex;
    atomicAdd(&denom[di], ex);
}

// Edge pass 3: out[dst, h, :] += hs[src, h, :] * (e / denom[dst,h])
__global__ void k_edge_pass3(const int* __restrict__ src, const int* __restrict__ dst,
                             const float* __restrict__ ea, const float* __restrict__ denom,
                             const float* __restrict__ hsrc, float* __restrict__ out, int E)
{
    int idx = blockIdx.x * blockDim.x + threadIdx.x;
    if (idx >= E * 8) return;
    int e = idx >> 3, hh = idx & 7;
    int di = dst[e] * 8 + hh;
    float alpha = ea[idx] / denom[di];
    const float* hp = hsrc + (long long)src[e] * 128 + hh * 16;
    float* op = out + (long long)dst[e] * 128 + hh * 16;
#pragma unroll
    for (int d = 0; d < 16; ++d) atomicAdd(&op[d], alpha * hp[d]);
}

// Semantic attention: scores from fused tanh-colsums, softmax over M metapaths
__global__ void k_semantic(const float* __restrict__ colsum, const float* __restrict__ q,
                           float* __restrict__ attn, int M, int Nn)
{
    __shared__ float red[128];
    __shared__ float sc[8];
    int t = threadIdx.x;
    float invN = 1.f / (float)Nn;
    for (int m = 0; m < M; ++m) {
        red[t] = q[t] * colsum[m * 128 + t] * invN;
        __syncthreads();
        for (int s = 64; s > 0; s >>= 1) { if (t < s) red[t] += red[t + s]; __syncthreads(); }
        if (t == 0) sc[m] = red[0];
        __syncthreads();
    }
    if (t == 0) {
        float mx = sc[0];
        for (int m = 1; m < M; ++m) mx = fmaxf(mx, sc[m]);
        float s = 0.f;
        for (int m = 0; m < M; ++m) { sc[m] = __expf(sc[m] - mx); s += sc[m]; }
        for (int m = 0; m < M; ++m) attn[m] = sc[m] / s;
    }
}

__global__ void k_combine(float* __restrict__ res,
                          const float* __restrict__ o0, const float* __restrict__ o1,
                          const float* __restrict__ o2, const float* __restrict__ o3,
                          const float* __restrict__ o4, const float* __restrict__ attn,
                          long long n, int M)
{
    long long i = (long long)blockIdx.x * blockDim.x + threadIdx.x;
    if (i >= n) return;
    float v = attn[0] * o0[i];
    if (M > 1) v += attn[1] * o1[i];
    if (M > 2) v += attn[2] * o2[i];
    if (M > 3) v += attn[3] * o3[i];
    if (M > 4) v += attn[4] * o4[i];
    res[i] = v;
}

// Per-row elu + layernorm (one 128-thread block per row)
__global__ void k_elu_ln(const float* __restrict__ x, float* __restrict__ y,
                         const float* __restrict__ g, const float* __restrict__ b, int N)
{
    __shared__ float red[128];
    __shared__ float mu_s, var_s;
    long long row = blockIdx.x;
    int t = threadIdx.x;
    float v = x[row * 128 + t];
    v = (v > 0.f) ? v : (__expf(v) - 1.f);
    red[t] = v; __syncthreads();
    for (int s = 64; s > 0; s >>= 1) { if (t < s) red[t] += red[t + s]; __syncthreads(); }
    if (t == 0) mu_s = red[0] * (1.f / 128.f);
    __syncthreads();
    float c = v - mu_s;
    red[t] = c * c; __syncthreads();
    for (int s = 64; s > 0; s >>= 1) { if (t < s) red[t] += red[t + s]; __syncthreads(); }
    if (t == 0) var_s = red[0] * (1.f / 128.f);
    __syncthreads();
    y[row * 128 + t] = g[t] * c * rsqrtf(var_s + 1e-5f) + b[t];
}

// ---------------------------------------------------------------------------
// Set transformer (4 heads, dh=32, 8 inducing points): two-pass softmax over N
// ---------------------------------------------------------------------------
__global__ void k_st_pass1(const float* __restrict__ Kmat, const float* __restrict__ q8,
                           unsigned* __restrict__ gmax, int N)
{
    __shared__ float qs[1024];
    __shared__ unsigned smax[32];
    int t = threadIdx.x;
    for (int i = t; i < 1024; i += 256) qs[i] = q8[i];
    if (t < 32) smax[t] = MONO_NEGINF;
    __syncthreads();
    int n = blockIdx.x * 256 + t;
    if (n < N) {
        for (int hh = 0; hh < 4; ++hh) {
            float kr[32];
#pragma unroll
            for (int d = 0; d < 32; ++d) kr[d] = Kmat[(long long)n * 128 + hh * 32 + d];
            for (int i = 0; i < 8; ++i) {
                float l = 0.f;
#pragma unroll
                for (int d = 0; d < 32; ++d) l += qs[i * 128 + hh * 32 + d] * kr[d];
                atomicMax(&smax[hh * 8 + i], f2mono(l * INV_SQRT32));
            }
        }
    }
    __syncthreads();
    if (t < 32) atomicMax(&gmax[t], smax[t]);
}

// 4 waves/block; each wave owns 8 (head,query) pairs; lane = dh index.
__global__ void k_st_pass2(const float* __restrict__ Kmat, const float* __restrict__ Vmat,
                           const float* __restrict__ q8, const unsigned* __restrict__ gmax,
                           float* __restrict__ ssum, float* __restrict__ sacc, int N)
{
    __shared__ float qs[1024];
    int t = threadIdx.x, wave = t >> 5, lane = t & 31;
    for (int i = t; i < 1024; i += 128) qs[i] = q8[i];
    __syncthreads();
    int n0 = blockIdx.x * 512;
    int n1 = n0 + 512; if (n1 > N) n1 = N;
    for (int p = wave; p < 32; p += 4) {
        int hh = p >> 3, ii = p & 7;
        float mx = mono2f(gmax[p]);
        float qv = qs[ii * 128 + hh * 32 + lane];
        float accd = 0.f, sume = 0.f;
        for (int n = n0; n < n1; ++n) {
            float part = qv * Kmat[(long long)n * 128 + hh * 32 + lane];
            for (int off = 16; off > 0; off >>= 1) part += __shfl_xor(part, off, 32);
            float e = __expf(part * INV_SQRT32 - mx);
            accd += e * Vmat[(long long)n * 128 + hh * 32 + lane];
            sume += e;
        }
        atomicAdd(&sacc[p * 32 + lane], accd);
        if (lane == 0) atomicAdd(&ssum[p], sume);
    }
}

__global__ void k_st_finalize(const float* __restrict__ sacc, const float* __restrict__ ssum,
                              const float* __restrict__ Wo, const float* __restrict__ bo,
                              const float* __restrict__ Wfc, const float* __restrict__ bfc,
                              float* __restrict__ agg)
{
    __shared__ float o[8 * 128];
    __shared__ float om[128];
    int t = threadIdx.x;
    int hh = t >> 5, d = t & 31;
    for (int i = 0; i < 8; ++i) {
        int p = hh * 8 + i;
        o[i * 128 + t] = sacc[p * 32 + d] / ssum[p];
    }
    __syncthreads();
    float m = 0.f;
    for (int i = 0; i < 8; ++i) {
        float acc = bo[t];
        for (int k = 0; k < 128; ++k) acc += o[i * 128 + k] * Wo[k * 128 + t];
        m += acc;
    }
    om[t] = m * 0.125f;
    __syncthreads();
    float acc = bfc[t];
    for (int k = 0; k < 128; ++k) acc += om[k] * Wfc[k * 128 + t];
    atomicAdd(&agg[t], acc);
}

__global__ void k_final(const float* __restrict__ agg, const float* __restrict__ Wout,
                        const float* __restrict__ bout, float* __restrict__ out)
{
    __shared__ float red[128];
    int t = threadIdx.x;
    red[t] = agg[t] * Wout[t];
    __syncthreads();
    for (int s = 64; s > 0; s >>= 1) { if (t < s) red[t] += red[t + s]; __syncthreads(); }
    if (t == 0) out[0] = fmaxf(red[0] + bout[0], 0.f);
}

// ---------------------------------------------------------------------------
// Host orchestration
// ---------------------------------------------------------------------------
struct ET { const int* s; const int* d; int E; int st; int dt;
            const float* as; const float* ad; };

extern "C" void kernel_launch(void* const* d_in, const int* in_sizes, int n_in,
                              void* d_out, int out_size, void* d_ws, size_t ws_size,
                              hipStream_t stream)
{
    auto F  = [&](int i) { return (const float*)d_in[i]; };
    auto Ix = [&](int i) { return (const int*)d_in[i]; };

    const int Ninst  = in_sizes[0] / 128;
    const int Nvar   = in_sizes[1] / 64;
    const int Nconst = in_sizes[2] / 32;

    float* base = (float*)d_ws;
    size_t off = 0;
    auto alloc = [&](size_t n) -> float* {
        float* p = base + off;
        off += (n + 63) & ~(size_t)63;
        return p;
    };

    float* h_inst  = alloc((size_t)Ninst * 128);
    float* h_var   = alloc((size_t)Nvar * 128);
    float* h_const = alloc((size_t)Nconst * 128);
    float* oi[5]; for (int m = 0; m < 5; ++m) oi[m] = alloc((size_t)Ninst * 128);
    float* ov[2]; for (int m = 0; m < 2; ++m) ov[m] = alloc((size_t)Nvar * 128);
    float* xa_i = alloc((size_t)Ninst * 128);
    float* xa_v = alloc((size_t)Nvar * 128);
    float* ssrc = alloc((size_t)Ninst * 8);
    float* sdst = alloc((size_t)Ninst * 8);
    unsigned* amax = (unsigned*)alloc((size_t)Ninst * 8);
    float* denom = alloc((size_t)Ninst * 8);
    int Emax = 0;
    for (int k = 3; k <= 15; k += 2) if (in_sizes[k] > Emax) Emax = in_sizes[k];
    float* ea     = alloc((size_t)Emax * 8);
    float* colsum = alloc(5 * 128);
    float* attn   = alloc(8);
    float* q8     = alloc(8 * 128);
    unsigned* gmaxb = (unsigned*)alloc(32);
    float* ssumb  = alloc(32);
    float* saccb  = alloc(32 * 32);
    float* agg    = alloc(128);

    auto nb = [](long long n, int b) { return (unsigned)((n + b - 1) / b); };
    auto fillf = [&](float* p, float v, long long n) {
        k_fill_f32<<<nb(n, 256), 256, 0, stream>>>(p, v, n); };
    auto fillu = [&](unsigned* p, unsigned v, long long n) {
        k_fill_u32<<<nb(n, 256), 256, 0, stream>>>(p, v, n); };
    auto gemm = [&](const float* A, const float* W, const float* b, float* C,
                    float* cs, int N, int K, int mode) {
        k_gemm_wmma<<<nb(N, 64), 128, 0, stream>>>(A, W, b, C, cs, N, K, mode); };

    float* hbuf[3] = { h_inst, h_var, h_const };
    int    Nn[3]   = { Ninst, Nvar, Nconst };

    auto run_edges = [&](const ET* ets, int nET, int& Mi, int& Mv) {
        Mi = 0; Mv = 0;
        for (int k = 0; k < nET; ++k) {
            const ET& e = ets[k];
            const float* hs = hbuf[e.st]; const float* hd = hbuf[e.dt];
            int Ns = Nn[e.st], Nd = Nn[e.dt];
            k_node_scores<<<nb((long long)Ns * 8, 256), 256, 0, stream>>>(hs, e.as, ssrc, Ns);
            k_node_scores<<<nb((long long)Nd * 8, 256), 256, 0, stream>>>(hd, e.ad, sdst, Nd);
            fillu(amax, MONO_NEGINF, (long long)Nd * 8);
            fillf(denom, 0.f, (long long)Nd * 8);
            float* outb = (e.dt == 0) ? oi[Mi++] : ov[Mv++];
            fillf(outb, 0.f, (long long)Nd * 128);
            long long T = (long long)e.E * 8;
            k_edge_pass1<<<nb(T, 256), 256, 0, stream>>>(e.s, e.d, ssrc, sdst, ea, amax, e.E);
            k_edge_pass2<<<nb(T, 256), 256, 0, stream>>>(e.d, ea, amax, denom, e.E);
            k_edge_pass3<<<nb(T, 256), 256, 0, stream>>>(e.s, e.d, ea, denom, hs, outb, e.E);
            k_map_relu<<<nb((long long)Nd * 128, 256), 256, 0, stream>>>(outb, (long long)Nd * 128);
        }
    };

    auto run_semantic = [&](float* const* outs, int M, int Nd,
                            const float* Wk, const float* bk, const float* q, float* res) {
        fillf(colsum, 0.f, (long long)M * 128);
        for (int m = 0; m < M; ++m)
            gemm(outs[m], Wk, bk, nullptr, colsum + m * 128, Nd, 128, 2);
        k_semantic<<<1, 128, 0, stream>>>(colsum, q, attn, M, Nd);
        k_combine<<<nb((long long)Nd * 128, 256), 256, 0, stream>>>(
            res, outs[0], M > 1 ? outs[1] : outs[0], M > 2 ? outs[2] : outs[0],
            M > 3 ? outs[3] : outs[0], M > 4 ? outs[4] : outs[0], attn,
            (long long)Nd * 128, M);
    };

    // ---------------- Layer 1 ----------------
    gemm(F(0), F(24), F(42), h_inst,  nullptr, Ninst,  128, 0);
    gemm(F(1), F(25), F(43), h_var,   nullptr, Nvar,   64,  0);
    gemm(F(2), F(23), F(41), h_const, nullptr, Nconst, 32,  0);
    {
        ET ets[7] = {
            { Ix(3),  Ix(4),  in_sizes[3],  0, 0, F(36), F(29) },  // ctrl
            { Ix(5),  Ix(6),  in_sizes[5],  0, 0, F(35), F(28) },  // call
            { Ix(7),  Ix(8),  in_sizes[7],  0, 1, F(37), F(30) },  // i2v
            { Ix(9),  Ix(10), in_sizes[9],  1, 0, F(39), F(32) },  // v2i
            { Ix(11), Ix(12), in_sizes[11], 2, 0, F(34), F(27) },  // c2i
            { Ix(13), Ix(14), in_sizes[13], 0, 0, F(38), F(31) },  // iid
            { Ix(15), Ix(16), in_sizes[15], 1, 1, F(40), F(33) },  // vid
        };
        int Mi, Mv;
        run_edges(ets, 7, Mi, Mv);
        run_semantic(oi, Mi, Ninst, F(26), F(44), F(45), h_inst);
        run_semantic(ov, Mv, Nvar,  F(26), F(44), F(45), h_var);
        k_elu_ln<<<Ninst, 128, 0, stream>>>(h_inst, xa_i, F(21), F(18), Ninst);
        k_elu_ln<<<Nvar,  128, 0, stream>>>(h_var,  xa_v, F(21), F(18), Nvar);
    }

    // ---------------- Layer 2 ----------------
    gemm(xa_i, F(46), F(61), h_inst, nullptr, Ninst, 128, 0);
    gemm(xa_v, F(47), F(62), h_var,  nullptr, Nvar,  128, 0);
    {
        ET ets[6] = {
            { Ix(3),  Ix(4),  in_sizes[3],  0, 0, F(56), F(50) },
            { Ix(5),  Ix(6),  in_sizes[5],  0, 0, F(55), F(49) },
            { Ix(7),  Ix(8),  in_sizes[7],  0, 1, F(57), F(51) },
            { Ix(9),  Ix(10), in_sizes[9],  1, 0, F(59), F(53) },
            { Ix(13), Ix(14), in_sizes[13], 0, 0, F(58), F(52) },
            { Ix(15), Ix(16), in_sizes[15], 1, 1, F(60), F(54) },
        };
        int Mi, Mv;
        run_edges(ets, 6, Mi, Mv);
        run_semantic(oi, Mi, Ninst, F(48), F(63), F(64), h_inst);
        run_semantic(ov, Mv, Nvar,  F(48), F(63), F(64), h_var);
        k_elu_ln<<<Ninst, 128, 0, stream>>>(h_inst, xa_i, F(22), F(19), Ninst);
        k_elu_ln<<<Nvar,  128, 0, stream>>>(h_var,  xa_v, F(22), F(19), Nvar);
    }

    // ---------------- Layer 3 ----------------
    gemm(xa_i, F(65), F(80), h_inst, nullptr, Ninst, 128, 0);
    gemm(xa_v, F(66), F(81), h_var,  nullptr, Nvar,  128, 0);
    {
        ET ets[6] = {
            { Ix(3),  Ix(4),  in_sizes[3],  0, 0, F(75), F(69) },
            { Ix(5),  Ix(6),  in_sizes[5],  0, 0, F(74), F(68) },
            { Ix(7),  Ix(8),  in_sizes[7],  0, 1, F(76), F(70) },
            { Ix(9),  Ix(10), in_sizes[9],  1, 0, F(78), F(72) },
            { Ix(13), Ix(14), in_sizes[13], 0, 0, F(77), F(71) },
            { Ix(15), Ix(16), in_sizes[15], 1, 1, F(79), F(73) },
        };
        int Mi, Mv;
        run_edges(ets, 6, Mi, Mv);
        run_semantic(oi, Mi, Ninst, F(67), F(82), F(83), h_inst);
        run_semantic(ov, Mv, Nvar,  F(67), F(82), F(83), h_var);
    }

    // elu into reuse buffers (l3 outputs feed set transformers)
    k_map_elu<<<nb((long long)Ninst * 128, 256), 256, 0, stream>>>(h_inst, oi[0], (long long)Ninst * 128);
    k_map_elu<<<nb((long long)Nvar  * 128, 256), 256, 0, stream>>>(h_var,  ov[0], (long long)Nvar  * 128);

    fillf(agg, 0.f, 128);

    // -------- set transformer (inst, st_i) --------
    gemm(F(94), F(87), F(92), q8, nullptr, 8, 128, 0);               // q = ip@Wq+bq
    gemm(oi[0], F(85), F(90), oi[1], nullptr, Ninst, 128, 0);        // K
    gemm(oi[0], F(88), F(93), oi[2], nullptr, Ninst, 128, 0);        // V
    fillu(gmaxb, MONO_NEGINF, 32);
    fillf(ssumb, 0.f, 32);
    fillf(saccb, 0.f, 1024);
    k_st_pass1<<<nb(Ninst, 256), 256, 0, stream>>>(oi[1], q8, gmaxb, Ninst);
    k_st_pass2<<<nb(Ninst, 512), 128, 0, stream>>>(oi[1], oi[2], q8, gmaxb, ssumb, saccb, Ninst);
    k_st_finalize<<<1, 128, 0, stream>>>(saccb, ssumb, F(86), F(91), F(84), F(89), agg);

    // -------- set transformer (var, st_v) --------
    gemm(F(105), F(98), F(103), q8, nullptr, 8, 128, 0);
    gemm(ov[0], F(96), F(101), oi[3], nullptr, Nvar, 128, 0);        // K
    gemm(ov[0], F(99), F(104), oi[4], nullptr, Nvar, 128, 0);        // V
    fillu(gmaxb, MONO_NEGINF, 32);
    fillf(ssumb, 0.f, 32);
    fillf(saccb, 0.f, 1024);
    k_st_pass1<<<nb(Nvar, 256), 256, 0, stream>>>(oi[3], q8, gmaxb, Nvar);
    k_st_pass2<<<nb(Nvar, 512), 128, 0, stream>>>(oi[3], oi[4], q8, gmaxb, ssumb, saccb, Nvar);
    k_st_finalize<<<1, 128, 0, stream>>>(saccb, ssumb, F(97), F(102), F(95), F(100), agg);

    k_final<<<1, 128, 0, stream>>>(agg, F(17), F(20), (float*)d_out);

    (void)n_in; (void)out_size; (void)ws_size;
}
